// ClassicalRBFKernel_65481071396529
// MI455X (gfx1250) — matrix operand order repaired and checked by reading, not compiled, not run on hardware
//
#include <hip/hip_runtime.h>

#define N_ROWS 8192
#define M_ROWS 8192
#define DDIM   512
#define RSTRIDE (2 * DDIM)   // interleaved hi|lo row stride (elements)

typedef __attribute__((ext_vector_type(8)))  __bf16 v8bf;
typedef __attribute__((ext_vector_type(16))) __bf16 v16bf;
typedef __attribute__((ext_vector_type(8)))  float  v8f;

// -------- Pass 1: exact fp32 row norms + interleaved bf16 hi/lo split ------
// Layout: pk[row*RSTRIDE + 0..511] = hi(row), pk[row*RSTRIDE + 512..1023] = lo(row)
__global__ __launch_bounds__(256)
void rbf_prep(const float* __restrict__ in,
              __bf16* __restrict__ pk, float* __restrict__ nrm)
{
    __shared__ float red[256];
    const int    row  = blockIdx.x;
    const int    t    = threadIdx.x;
    const size_t ibase = (size_t)row * DDIM;
    const size_t obase = (size_t)row * RSTRIDE;
    float s = 0.0f;
#pragma unroll
    for (int i = 0; i < DDIM / 256; ++i) {
        const int idx = t + i * 256;
        float v = in[ibase + idx];
        __bf16 h = (__bf16)v;              // RNE to bf16
        float  l = v - (float)h;           // residual fits in bf16
        pk[obase + idx]        = h;
        pk[obase + DDIM + idx] = (__bf16)l;
        s += v * v;                        // exact fp32 norm
    }
    red[t] = s;
    __syncthreads();
#pragma unroll
    for (int off = 128; off > 0; off >>= 1) {
        if (t < off) red[t] += red[t + off];
        __syncthreads();
    }
    if (t == 0) nrm[row] = red[0];
}

// -------- Pass 2: fused 3-term bf16 WMMA GEMM + RBF epilogue ---------------
// Block: 256 threads (8 waves), tile 128x128. Wave tile 64x32 (4x2 of 16x16).
__global__ __launch_bounds__(256)
void rbf_wmma_main(const __bf16* __restrict__ xp, const __bf16* __restrict__ yp,
                   const float* __restrict__ x2, const float* __restrict__ y2,
                   const float* __restrict__ gamma_p, float* __restrict__ out)
{
    const int lane = threadIdx.x & 31;
    const int w    = threadIdx.x >> 5;
    const int wr   = w & 1;          // 2 wave rows
    const int wc   = w >> 1;         // 4 wave cols
    const int rbase = blockIdx.y * 128 + wr * 64;
    const int cbase = blockIdx.x * 128 + wc * 32;

    const int mrow  = lane & 15;     // M (A) or N (B) index within tile
    const int khalf = lane >> 4;     // which K-half this lane holds

    v8f acc[4][2];
#pragma unroll
    for (int t = 0; t < 4; ++t)
#pragma unroll
        for (int u = 0; u < 2; ++u)
            acc[t][u] = (v8f){0.f, 0.f, 0.f, 0.f, 0.f, 0.f, 0.f, 0.f};

    // One per-lane base pointer per fragment row/col; hi/lo and K-chunk
    // halves become small immediate offsets (+8, +512, +520 elements).
    const __bf16* apos[4];
    const __bf16* bpos[2];
#pragma unroll
    for (int t = 0; t < 4; ++t)
        apos[t] = xp + (size_t)(rbase + t * 16 + mrow) * RSTRIDE + khalf * 16;
#pragma unroll
    for (int u = 0; u < 2; ++u)
        bpos[u] = yp + (size_t)(cbase + u * 16 + mrow) * RSTRIDE + khalf * 16;

    for (int k0 = 0; k0 < DDIM; k0 += 32) {
        v16bf ah[4], al[4], bh[2], bl[2];
        // A fragment (16x32 bf16): lane<16 -> K[0..15], lane>=16 -> K[16..31]
#pragma unroll
        for (int t = 0; t < 4; ++t) {
            const __bf16* p = apos[t] + k0;
            v8bf h0 = *(const v8bf*)(p);
            v8bf h1 = *(const v8bf*)(p + 8);
            v8bf l0 = *(const v8bf*)(p + DDIM);
            v8bf l1 = *(const v8bf*)(p + DDIM + 8);
            ah[t] = __builtin_shufflevector(h0, h1, 0,1,2,3,4,5,6,7,8,9,10,11,12,13,14,15);
            al[t] = __builtin_shufflevector(l0, l1, 0,1,2,3,4,5,6,7,8,9,10,11,12,13,14,15);
        }
        // B fragment (32x16): lane holds y[col, k0 + khalf*16 .. +15] contiguous
#pragma unroll
        for (int u = 0; u < 2; ++u) {
            const __bf16* p = bpos[u] + k0;
            bh[u] = *(const v16bf*)(p);
            bl[u] = *(const v16bf*)(p + DDIM);
        }
        // 3-term product: xh*yh + xh*yl + xl*yh (lo*lo dropped)
#pragma unroll
        for (int t = 0; t < 4; ++t)
#pragma unroll
            for (int u = 0; u < 2; ++u) {
                acc[t][u] = __builtin_amdgcn_wmma_f32_16x16x32_bf16(
                    false, ah[t], false, bh[u], (short)0, acc[t][u], false, false);
                acc[t][u] = __builtin_amdgcn_wmma_f32_16x16x32_bf16(
                    false, ah[t], false, bl[u], (short)0, acc[t][u], false, false);
                acc[t][u] = __builtin_amdgcn_wmma_f32_16x16x32_bf16(
                    false, al[t], false, bh[u], (short)0, acc[t][u], false, false);
            }
    }

    // Epilogue: exp(-gamma * max(0, x2 + y2 - 2*xy)); C/D layout:
    // lane<16: VGPR j -> M=j ; lane>=16: VGPR j -> M=8+j ; N = lane&15.
    // Output is write-once (never re-read): non-temporal stores keep the
    // bf16 operand tiles resident in the 192MB L2.
    const float g = *gamma_p;
#pragma unroll
    for (int t = 0; t < 4; ++t) {
        const int r0 = rbase + t * 16 + khalf * 8;
        float xr[8];
#pragma unroll
        for (int j = 0; j < 8; ++j) xr[j] = x2[r0 + j];
#pragma unroll
        for (int u = 0; u < 2; ++u) {
            const int   col = cbase + u * 16 + mrow;
            const float yc  = y2[col];
#pragma unroll
            for (int j = 0; j < 8; ++j) {
                float sq = xr[j] + yc - 2.0f * acc[t][u][j];
                sq = fmaxf(sq, 0.0f);
                __builtin_nontemporal_store(__expf(-g * sq),
                                            out + (size_t)(r0 + j) * M_ROWS + col);
            }
        }
    }
}

// ---------------------------------------------------------------------------
extern "C" void kernel_launch(void* const* d_in, const int* in_sizes, int n_in,
                              void* d_out, int out_size, void* d_ws, size_t ws_size,
                              hipStream_t stream)
{
    const float* x       = (const float*)d_in[0];
    const float* y       = (const float*)d_in[1];
    const float* gamma_p = (const float*)d_in[2];
    float* out = (float*)d_out;

    // Workspace: xp (N*2D bf16, 16MB) | yp (M*2D bf16, 16MB) | x2 | y2
    char* ws = (char*)d_ws;
    __bf16* xp = (__bf16*)ws;
    __bf16* yp = xp + (size_t)N_ROWS * RSTRIDE;
    float*  x2 = (float*)(yp + (size_t)M_ROWS * RSTRIDE);
    float*  y2 = x2 + N_ROWS;

    rbf_prep<<<N_ROWS, 256, 0, stream>>>(x, xp, x2);
    rbf_prep<<<M_ROWS, 256, 0, stream>>>(y, yp, y2);

    dim3 grid(M_ROWS / 128, N_ROWS / 128);
    rbf_wmma_main<<<grid, 256, 0, stream>>>(xp, yp, x2, y2, gamma_p, out);
}
